// GQA_37366215475614
// MI455X (gfx1250) — matrix operand order repaired and checked by reading, not compiled, not run on hardware
//
#include <hip/hip_runtime.h>
#include <hip/hip_bf16.h>

// ---------------------------------------------------------------------------
// GQA attention block for gfx1250 (MI455X).  B=2, S=1024, D=2048, H=32,
// HKV=8, dk=64, G=4.  All GEMM-like stages run on v_wmma_f32_16x16x32_bf16.
// This revision: softmax reductions moved off the DS unit --
//   * row-sum computed by an extra WMMA against an all-ones B fragment
//   * row-max reduced with DPP (quad_perm / row_half_mirror / row_mirror)
// ---------------------------------------------------------------------------

#define B_   2
#define S_   1024
#define D_   2048
#define H_   32
#define HKV_ 8
#define DK_  64
#define R_   (B_ * S_)       // 2048 rows of x / q / k / v
#define DKV_ (HKV_ * DK_)    // 512

typedef __attribute__((ext_vector_type(16))) __bf16 v16bf;
typedef __attribute__((ext_vector_type(8)))  __bf16 v8bf;
typedef __attribute__((ext_vector_type(8)))  float  v8f;
typedef unsigned short ushort_t;

union FragBF {
    v16bf    v;
    v8bf     h[2];
    ushort_t u[16];
};

__device__ __forceinline__ float bf2f(ushort_t u) {
    unsigned int x = ((unsigned int)u) << 16;
    return __builtin_bit_cast(float, x);
}
__device__ __forceinline__ ushort_t f2bf(float f) {
    unsigned int u = __builtin_bit_cast(unsigned int, f);
    unsigned int r = u + 0x7FFFu + ((u >> 16) & 1u);   // round-nearest-even
    return (ushort_t)(r >> 16);
}

__device__ __forceinline__ v8f wmma_bf16(const FragBF& a, const FragBF& b, v8f c) {
    // D = A(16x32 bf16) * B(32x16 bf16) + C(16x16 f32)
    return __builtin_amdgcn_wmma_f32_16x16x32_bf16(
        /*neg_a=*/false, a.v, /*neg_b=*/false, b.v,
        /*c_mod=*/(short)0, c, /*reuse_a=*/false, /*reuse_b=*/false);
}

// one DPP max step (VALU only; no LDS)
template <int CTRL>
__device__ __forceinline__ float fmax_dpp(float x) {
    int yi = __builtin_amdgcn_mov_dpp(__builtin_bit_cast(int, x),
                                      CTRL, 0xF, 0xF, true);
    return fmaxf(x, __builtin_bit_cast(float, yi));
}
// max over each 16-lane half of the wave (rows of the C fragment)
__device__ __forceinline__ float rowmax16_dpp(float x) {
    x = fmax_dpp<0x0B1>(x);   // quad_perm [1,0,3,2]  (xor 1)
    x = fmax_dpp<0x04E>(x);   // quad_perm [2,3,0,1]  (xor 2)
    x = fmax_dpp<0x141>(x);   // row_half_mirror      (xor 4 after 1,2)
    x = fmax_dpp<0x140>(x);   // row_mirror           (xor 8 after 1,2,4)
    return x;
}

// async copy of 32 contiguous bytes global -> LDS (per lane), ASYNCcnt-tracked
__device__ __forceinline__ void async_copy32(unsigned lds_off,
                                             const ushort_t* src) {
    unsigned long long ga = (unsigned long long)(uintptr_t)src;
    asm volatile("global_load_async_to_lds_b128 %0, %1, off"
                 :: "v"(lds_off), "v"(ga) : "memory");
    asm volatile("global_load_async_to_lds_b128 %0, %1, off offset:16"
                 :: "v"(lds_off), "v"(ga) : "memory");
}

// ---------------------------------------------------------------------------
// Stage 0a: fp32 -> bf16 elementwise (for x)
// ---------------------------------------------------------------------------
__global__ void f32_to_bf16_kernel(const float* __restrict__ in,
                                   ushort_t* __restrict__ out, int n) {
    int i = blockIdx.x * blockDim.x + threadIdx.x;
    if (i < n) out[i] = f2bf(in[i]);
}

// ---------------------------------------------------------------------------
// Stage 0b: fp32 -> bf16 with transpose (for weights).
// src[Kd][Nd] row-major  ->  dst[Nd][Kd] row-major.  LDS 32x32 tile.
// ---------------------------------------------------------------------------
__global__ __launch_bounds__(256) void f32_to_bf16_transpose_kernel(
    const float* __restrict__ src, ushort_t* __restrict__ dst, int Kd, int Nd) {
    __shared__ float tile[32][33];
    const int k0 = blockIdx.y * 32;
    const int n0 = blockIdx.x * 32;
    const int t  = threadIdx.x;
    const int r  = t >> 3;            // 0..31
    const int c4 = (t & 7) * 4;       // 0,4,...,28

    const float* s = src + (size_t)(k0 + r) * Nd + n0 + c4;
    tile[r][c4 + 0] = s[0];
    tile[r][c4 + 1] = s[1];
    tile[r][c4 + 2] = s[2];
    tile[r][c4 + 3] = s[3];
    __syncthreads();

    union { ushort_t u[4]; unsigned long long q; } o;
    o.u[0] = f2bf(tile[c4 + 0][r]);
    o.u[1] = f2bf(tile[c4 + 1][r]);
    o.u[2] = f2bf(tile[c4 + 2][r]);
    o.u[3] = f2bf(tile[c4 + 3][r]);
    *(unsigned long long*)(dst + (size_t)(n0 + r) * Kd + k0 + c4) = o.q;
}

// ---------------------------------------------------------------------------
// Stage 1/4: WMMA GEMM with async-LDS double-buffered B staging.
//   A[M,K]  bf16 row-major,  Wt[N,K] bf16 row-major (pre-transposed weights).
//   Block = 8 waves stacked in M (128 x 64 tile); wave = 16(M) x 64(N).
//   K-step 64.  B tile (64 cols x 64 K = 8KB) staged once per block via
//   global_load_async_to_lds_b128, consumed by all 8 waves from LDS.
// OUT_MODE: 0 = bf16 row-major C[M,N]
//           1 = f32  row-major C[M,N]
//           2 = bf16 transposed C[N,M] (packed 16B stores; used for V)
// ---------------------------------------------------------------------------
template <int OUT_MODE>
__global__ __launch_bounds__(256) void gemm_wmma_kernel(
    const ushort_t* __restrict__ A, const ushort_t* __restrict__ Wt,
    void* __restrict__ Cv, int M, int N, int K) {
    __shared__ ushort_t Btile[2][64][72];   // 144B rows (16B aligned)

    const int t    = threadIdx.x;
    const int w    = t >> 5;
    const int lane = t & 31;
    const int lm   = lane & 15;
    const int hi   = lane >> 4;
    const int row0 = blockIdx.y * 128 + w * 16;
    const int col0 = blockIdx.x * 64;

    const ushort_t* arow = A + (size_t)(row0 + lm) * K;

    // async B staging: thread t copies 32B of row (col0 + t/4), kc = (t%4)*16
    const unsigned lds0 = (unsigned)(uintptr_t)(void*)&Btile[0][0][0];
    const int  sn  = t >> 2;
    const int  skc = (t & 3) * 16;
    const ushort_t* bsrc_row = Wt + (size_t)(col0 + sn) * K + skc;
    const unsigned  bdst_off = (unsigned)(sn * 144 + skc * 2);

    auto stage = [&](int k0, int buf) {
        async_copy32(lds0 + (unsigned)buf * (64u * 144u) + bdst_off,
                     bsrc_row + k0);
    };

    stage(0, 0);
    stage(64, 1);

    v8f acc[4] = {};
    for (int k0 = 0; k0 < K; k0 += 64) {
        const int buf = (k0 >> 6) & 1;
        // complete the stage for `buf` (async loads finish in order)
        if (k0 + 64 < K)
            asm volatile("s_wait_asynccnt 0x2" ::: "memory");
        else
            asm volatile("s_wait_asynccnt 0x0" ::: "memory");
        __syncthreads();

        if (k0 + 64 < K)
            __builtin_prefetch(arow + k0 + 64, 0, 1);   // global_prefetch_b8

        // A fragments direct from global (contiguous 16B chunks)
        FragBF a0, a1;
        a0.h[0] = *(const v8bf*)(arow + k0 + hi * 16);
        a0.h[1] = *(const v8bf*)(arow + k0 + hi * 16 + 8);
        a1.h[0] = *(const v8bf*)(arow + k0 + 32 + hi * 16);
        a1.h[1] = *(const v8bf*)(arow + k0 + 32 + hi * 16 + 8);
#pragma unroll
        for (int nn = 0; nn < 4; ++nn) {
            const ushort_t* brow = &Btile[buf][nn * 16 + lm][0];
            FragBF b;
            b.h[0] = *(const v8bf*)(brow + hi * 16);
            b.h[1] = *(const v8bf*)(brow + hi * 16 + 8);
            acc[nn] = wmma_bf16(a0, b, acc[nn]);
            b.h[0] = *(const v8bf*)(brow + 32 + hi * 16);
            b.h[1] = *(const v8bf*)(brow + 32 + hi * 16 + 8);
            acc[nn] = wmma_bf16(a1, b, acc[nn]);
        }

        __syncthreads();                       // all waves done reading `buf`
        if (k0 + 128 < K) stage(k0 + 128, buf);
    }

#pragma unroll
    for (int nn = 0; nn < 4; ++nn) {
        const int col = col0 + nn * 16 + lm;
        if (OUT_MODE == 2) {
            // transposed write: lane's 8 rows are contiguous -> one 16B store
            union { ushort_t u[8]; uint4 q; } o;
#pragma unroll
            for (int r = 0; r < 8; ++r) o.u[r] = f2bf(acc[nn][r]);
            *(uint4*)((ushort_t*)Cv + (size_t)col * M + row0 + hi * 8) = o.q;
        } else {
#pragma unroll
            for (int r = 0; r < 8; ++r) {
                const int row = row0 + hi * 8 + r;
                if (OUT_MODE == 1)
                    ((float*)Cv)[(size_t)row * N + col] = acc[nn][r];
                else
                    ((ushort_t*)Cv)[(size_t)row * N + col] = f2bf(acc[nn][r]);
            }
        }
    }
}

// ---------------------------------------------------------------------------
// Stage 2: interleaved-pair RoPE (in place on bf16).  scale folds 1/sqrt(dk)
// into Q.  t: [rows, nh*64], pos = row % S.
// ---------------------------------------------------------------------------
__global__ void rope_kernel(ushort_t* __restrict__ t, int rows, int nh,
                            float scale) {
    int i = blockIdx.x * blockDim.x + threadIdx.x;
    int total = rows * nh * (DK_ / 2);
    if (i >= total) return;
    int p   = i & 31;                 // pair index 0..31
    int h   = (i >> 5) % nh;
    int row = i / (32 * nh);
    int pos = row % S_;

    float freq = __powf(10000.0f, -2.0f * (float)p / (float)DK_);
    float sn, cs;
    __sincosf((float)pos * freq, &sn, &cs);

    ushort_t* e = t + (size_t)row * (nh * DK_) + h * DK_ + 2 * p;
    float t0 = bf2f(e[0]);
    float t1 = bf2f(e[1]);
    e[0] = f2bf((t0 * cs - t1 * sn) * scale);
    e[1] = f2bf((t0 * sn + t1 * cs) * scale);
}

// ---------------------------------------------------------------------------
// Stage 3: causal flash attention.  One wave per (batch, head, 16-query tile).
// Key tile = 32.  MASKED constant-folds away in the hot loop.  Row-max via
// DPP; row-sum via WMMA against an all-ones fragment (accumulator lacc).
// ---------------------------------------------------------------------------
template <bool MASKED>
__device__ __forceinline__ void attn_keytile(
    int key0, int qt, int lm, int hi,
    const ushort_t* __restrict__ kbase,   // K rows base for (b, kvh)
    const ushort_t* __restrict__ vbase,   // V^T base for (b, kvh)
    ushort_t (*__restrict__ Pw)[32],      // per-wave P tile [16][32]
    const FragBF& aq0, const FragBF& aq1, const FragBF& ones,
    v8f (&oacc)[4], v8f& lacc, float (&mrow)[8], bool pf) {
    // ---- scores: S = Q (16x64) . K^T (64x32), 2 N-tiles x 2 K-halves ----
    const ushort_t* kr0 = kbase + (size_t)(key0 + lm) * DKV_;
    const ushort_t* kr1 = kbase + (size_t)(key0 + 16 + lm) * DKV_;
    if (pf) {
        __builtin_prefetch(kr0 + 32 * DKV_, 0, 1);   // next key tile
        __builtin_prefetch(kr1 + 32 * DKV_, 0, 1);
    }
    v8f s0 = {}, s1 = {};
    FragBF kb;
    kb.h[0] = *(const v8bf*)(kr0 + hi * 16);
    kb.h[1] = *(const v8bf*)(kr0 + hi * 16 + 8);
    s0 = wmma_bf16(aq0, kb, s0);
    kb.h[0] = *(const v8bf*)(kr0 + 32 + hi * 16);
    kb.h[1] = *(const v8bf*)(kr0 + 32 + hi * 16 + 8);
    s0 = wmma_bf16(aq1, kb, s0);
    kb.h[0] = *(const v8bf*)(kr1 + hi * 16);
    kb.h[1] = *(const v8bf*)(kr1 + hi * 16 + 8);
    s1 = wmma_bf16(aq0, kb, s1);
    kb.h[0] = *(const v8bf*)(kr1 + 32 + hi * 16);
    kb.h[1] = *(const v8bf*)(kr1 + 32 + hi * 16 + 8);
    s1 = wmma_bf16(aq1, kb, s1);

    // ---- online softmax over the 16x32 score block ----
#pragma unroll
    for (int r = 0; r < 8; ++r) {
        const int Mr = hi * 8 + r;
        float e0, e1;
        if (MASKED) {
            const int qglob = qt * 16 + Mr;
            e0 = (key0 + lm      <= qglob) ? s0[r] : -3.0e38f;
            e1 = (key0 + 16 + lm <= qglob) ? s1[r] : -3.0e38f;
        } else {
            e0 = s0[r];
            e1 = s1[r];
        }
        float mx   = rowmax16_dpp(fmaxf(e0, e1));     // VALU-only reduction
        float mnew = fmaxf(mrow[r], mx);
        float corr = __expf(mrow[r] - mnew);
        float p0   = __expf(e0 - mnew);
        float p1   = __expf(e1 - mnew);
        mrow[r] = mnew;
        oacc[0][r] *= corr; oacc[1][r] *= corr;
        oacc[2][r] *= corr; oacc[3][r] *= corr;
        lacc[r]    *= corr;
        Pw[Mr][lm]      = f2bf(p0);
        Pw[Mr][16 + lm] = f2bf(p1);
    }

    // ---- O += P (16x32) . V (32x64); row-sums += P . ones (WMMA) ----
    FragBF pa;
    pa.h[0] = *(const v8bf*)&Pw[lm][hi * 16];
    pa.h[1] = *(const v8bf*)&Pw[lm][hi * 16 + 8];
#pragma unroll
    for (int nn = 0; nn < 4; ++nn) {
        const ushort_t* vr = vbase + (size_t)(nn * 16 + lm) * R_ + key0;
        FragBF vb;
        vb.h[0] = *(const v8bf*)(vr + hi * 16);
        vb.h[1] = *(const v8bf*)(vr + hi * 16 + 8);
        oacc[nn] = wmma_bf16(pa, vb, oacc[nn]);
    }
    lacc = wmma_bf16(pa, ones, lacc);   // every lane gets its row's sum
}

__global__ __launch_bounds__(256) void flash_attn_kernel(
    const ushort_t* __restrict__ Q, const ushort_t* __restrict__ Kc,
    const ushort_t* __restrict__ Vt, ushort_t* __restrict__ O) {
    __shared__ ushort_t Pbuf[8][16][32];   // per-wave P tile (64B rows)

    const int w    = threadIdx.x >> 5;
    const int lane = threadIdx.x & 31;
    const int lm   = lane & 15;
    const int hi   = lane >> 4;

    const int task = blockIdx.x * 8 + w;           // 0 .. B*H*(S/16)-1
    const int qt   = task & 63;                    // query tile
    const int hh   = task >> 6;
    const int b    = hh >> 5;
    const int h    = hh & 31;
    const int kvh  = h >> 2;                       // GQA group

    // --- load Q tile once (A fragments, two dk halves), scale pre-folded ---
    const ushort_t* qrow =
        Q + (size_t)(b * S_ + qt * 16 + lm) * D_ + h * DK_;
    FragBF aq0, aq1;
    aq0.h[0] = *(const v8bf*)(qrow + hi * 16);
    aq0.h[1] = *(const v8bf*)(qrow + hi * 16 + 8);
    aq1.h[0] = *(const v8bf*)(qrow + 32 + hi * 16);
    aq1.h[1] = *(const v8bf*)(qrow + 32 + hi * 16 + 8);

    FragBF ones;
#pragma unroll
    for (int e = 0; e < 16; ++e) ones.u[e] = 0x3F80;   // bf16 1.0

    const ushort_t* kbase = Kc + (size_t)(b * S_) * DKV_ + kvh * DK_;
    const ushort_t* vbase = Vt + (size_t)(kvh * DK_) * R_ + b * S_;

    v8f oacc[4] = {};
    v8f lacc = {};
    float mrow[8];
#pragma unroll
    for (int r = 0; r < 8; ++r) mrow[r] = -3.0e38f;

    const int ktend  = qt >> 1;                       // last key tile
    const int ktfull = (qt * 16 - 31) / 32;           // last fully-unmasked
    int kt = 0;
    for (; kt <= ktfull; ++kt)
        attn_keytile<false>(kt * 32, qt, lm, hi, kbase, vbase, Pbuf[w],
                            aq0, aq1, ones, oacc, lacc, mrow, kt < ktend);
    for (; kt <= ktend; ++kt)
        attn_keytile<true>(kt * 32, qt, lm, hi, kbase, vbase, Pbuf[w],
                           aq0, aq1, ones, oacc, lacc, mrow, false);

    // ---- normalize and write attention output [B*S, H*dk] (bf16) ----
#pragma unroll
    for (int r = 0; r < 8; ++r) {
        const int Mr = hi * 8 + r;
        float inv = 1.0f / lacc[r];
        size_t base = (size_t)(b * S_ + qt * 16 + Mr) * D_ + h * DK_ + lm;
        O[base +  0] = f2bf(oacc[0][r] * inv);
        O[base + 16] = f2bf(oacc[1][r] * inv);
        O[base + 32] = f2bf(oacc[2][r] * inv);
        O[base + 48] = f2bf(oacc[3][r] * inv);
    }
}

// ---------------------------------------------------------------------------
// Host-side launch
// ---------------------------------------------------------------------------
extern "C" void kernel_launch(void* const* d_in, const int* in_sizes, int n_in,
                              void* d_out, int out_size, void* d_ws, size_t ws_size,
                              hipStream_t stream) {
    const float* x  = (const float*)d_in[0];
    const float* Wq = (const float*)d_in[1];
    const float* Wk = (const float*)d_in[2];
    const float* Wv = (const float*)d_in[3];
    const float* Wo = (const float*)d_in[4];
    float* out = (float*)d_out;

    // workspace carve-up (bf16 buffers), all 16B aligned
    char* ws = (char*)d_ws;
    ushort_t* xb  = (ushort_t*)ws; ws += (size_t)R_ * D_   * 2;  // x bf16
    ushort_t* Wqt = (ushort_t*)ws; ws += (size_t)D_ * D_   * 2;  // Wq^T
    ushort_t* Wkt = (ushort_t*)ws; ws += (size_t)D_ * DKV_ * 2;  // Wk^T
    ushort_t* Wvt = (ushort_t*)ws; ws += (size_t)D_ * DKV_ * 2;  // Wv^T
    ushort_t* Wot = (ushort_t*)ws; ws += (size_t)D_ * D_   * 2;  // Wo^T
    ushort_t* Qb  = (ushort_t*)ws; ws += (size_t)R_ * D_   * 2;  // Q rows
    ushort_t* Kb  = (ushort_t*)ws; ws += (size_t)R_ * DKV_ * 2;  // K rows
    ushort_t* Vtb = (ushort_t*)ws; ws += (size_t)R_ * DKV_ * 2;  // V transposed
    ushort_t* AOb = (ushort_t*)ws; ws += (size_t)R_ * D_   * 2;  // attn out

    // convert x (elementwise) and weights (transposing)
    {
        int n = R_ * D_;
        f32_to_bf16_kernel<<<(n + 255) / 256, 256, 0, stream>>>(x, xb, n);
    }
    f32_to_bf16_transpose_kernel<<<dim3(D_ / 32,   D_ / 32), 256, 0, stream>>>(
        Wq, Wqt, D_, D_);
    f32_to_bf16_transpose_kernel<<<dim3(DKV_ / 32, D_ / 32), 256, 0, stream>>>(
        Wk, Wkt, D_, DKV_);
    f32_to_bf16_transpose_kernel<<<dim3(DKV_ / 32, D_ / 32), 256, 0, stream>>>(
        Wv, Wvt, D_, DKV_);
    f32_to_bf16_transpose_kernel<<<dim3(D_ / 32,   D_ / 32), 256, 0, stream>>>(
        Wo, Wot, D_, D_);

    // QKV projections (Q,K bf16 row-major; V bf16 transposed)
    gemm_wmma_kernel<0><<<dim3(D_ / 64,   R_ / 128), 256, 0, stream>>>(
        xb, Wqt, (void*)Qb, R_, D_, D_);
    gemm_wmma_kernel<0><<<dim3(DKV_ / 64, R_ / 128), 256, 0, stream>>>(
        xb, Wkt, (void*)Kb, R_, DKV_, D_);
    gemm_wmma_kernel<2><<<dim3(DKV_ / 64, R_ / 128), 256, 0, stream>>>(
        xb, Wvt, (void*)Vtb, R_, DKV_, D_);

    // RoPE (fold 1/sqrt(dk)=0.125 into Q); V needs no RoPE
    {
        int nq = R_ * H_   * (DK_ / 2);
        int nk = R_ * HKV_ * (DK_ / 2);
        rope_kernel<<<(nq + 255) / 256, 256, 0, stream>>>(Qb, R_, H_,   0.125f);
        rope_kernel<<<(nk + 255) / 256, 256, 0, stream>>>(Kb, R_, HKV_, 1.0f);
    }

    // causal flash attention: B*H*(S/16) = 4096 wave-tasks, 8 waves/block
    flash_attn_kernel<<<(B_ * H_ * (S_ / 16)) / 8, 256, 0, stream>>>(
        Qb, Kb, Vtb, AOb);

    // output projection (fp32 out -> d_out)
    gemm_wmma_kernel<1><<<dim3(D_ / 64, R_ / 128), 256, 0, stream>>>(
        AOb, Wot, (void*)out, R_, D_, D_);
}
